// SpectralAttention_23390391894539
// MI455X (gfx1250) — compile-verified, matrix operand order checked by reading, not run on hardware
//
#include <hip/hip_runtime.h>
#include <hip/hip_bf16.h>

// ---------------------------------------------------------------------------
// Causal MHA block for MI455X (gfx1250, wave32, WMMA + TDM).
//   x[2,2048,1024] f32, w_qkv[3072,1024] f32, w_proj[1024,1024] f32
//   out[2,2048,1024] f32
// GEMMs: bf16 WMMA fed from LDS, tiles staged by the Tensor Data Mover
// (tensor_load_to_lds, TENSORcnt-tracked, double buffered).
// Attention: flash-style online softmax with WMMA for QK^T and PV.
// ---------------------------------------------------------------------------

#define B_   2
#define S_   2048
#define D_   1024
#define H_   16
#define DH_  64
#define MTOT (B_ * S_)   // 4096
#define E3   (3 * D_)    // 3072

typedef __attribute__((ext_vector_type(16))) __bf16 v16bf;
typedef __attribute__((ext_vector_type(8)))  float  v8f;
typedef unsigned int uint32x4 __attribute__((ext_vector_type(4)));
typedef int          int32x8  __attribute__((ext_vector_type(8)));
typedef int          int32x4  __attribute__((ext_vector_type(4)));

#if __has_builtin(__builtin_amdgcn_tensor_load_to_lds) && \
    __has_builtin(__builtin_amdgcn_s_wait_tensorcnt)
#define HAVE_TDM 1
#else
#define HAVE_TDM 0
#endif

static __device__ __forceinline__ v8f wmma_bf16(v16bf a, v16bf b, v8f c) {
  return __builtin_amdgcn_wmma_f32_16x16x32_bf16(false, a, false, b, (short)0, c,
                                                 false, false);
}

// K-major fragment load (A or B operand). Lane L supplies row/col (L & 15)
// and the (L >> 4)-th 16-element K half. Any consistent K-permutation applied
// to BOTH operands leaves the WMMA dot product invariant, so contiguous-K
// loads are correct for X*W^T-style products. Works for global and LDS.
static __device__ __forceinline__ v16bf load_frag(const __bf16* p, int ld, int lane) {
  return *(const v16bf*)(p + (size_t)(lane & 15) * (size_t)ld +
                         (size_t)((lane >> 4) << 4));
}

// Low 32 bits of a flat pointer into the LDS aperture = byte offset within the
// workgroup's LDS allocation (aperture check uses ADDR[63:32]).
static __device__ __forceinline__ unsigned lds_offset(const void* p) {
  return (unsigned)(unsigned long long)(uintptr_t)p;
}

#if HAVE_TDM
// Issue a 2D TDM load: tile (rows x 64 bf16) from a row-major tensor with
// row stride `ld` elements into contiguous LDS (row-major [rows][64]).
static __device__ __forceinline__ void tdm_load_tile(
    unsigned lds_off, const __bf16* gsrc, unsigned tile_rows,
    unsigned tensor_rows, unsigned ld_elts) {
  const unsigned long long ga = (unsigned long long)(uintptr_t)gsrc;
  uint32x4 g0;
  g0[0] = 1u;                                            // count=1, user desc
  g0[1] = lds_off;                                       // D#.lds_addr
  g0[2] = (unsigned)(ga & 0xffffffffull);                // global_addr[31:0]
  g0[3] = (unsigned)((ga >> 32) & 0x1ffffffull) | (2u << 30);  // [56:32]|type=2
  int32x8 g1;
  g1[0] = (int)(1u << 16);                               // data_size=2B, mask=0
  g1[1] = (int)((ld_elts & 0xffffu) << 16);              // tensor_dim0[15:0]
  g1[2] = (int)(((ld_elts >> 16) & 0xffffu) |            // tensor_dim0[31:16]
                ((tensor_rows & 0xffffu) << 16));        // tensor_dim1[15:0]
  g1[3] = (int)(((tensor_rows >> 16) & 0xffffu) |        // tensor_dim1[31:16]
                (64u << 16));                            // tile_dim0 = 64
  g1[4] = (int)(tile_rows & 0xffffu);                    // tile_dim1; dim2=0
  g1[5] = (int)ld_elts;                                  // dim0_stride[31:0]
  g1[6] = 0;                                             // strides hi = 0
  g1[7] = 0;
  int32x4 z4;  z4[0] = 0; z4[1] = 0; z4[2] = 0; z4[3] = 0;
#if __clang_major__ >= 23
  int32x8 z8;  z8[0] = 0; z8[1] = 0; z8[2] = 0; z8[3] = 0;
  z8[4] = 0; z8[5] = 0; z8[6] = 0; z8[7] = 0;
  __builtin_amdgcn_tensor_load_to_lds(g0, g1, z4, z4, z8, 0);
#else
  __builtin_amdgcn_tensor_load_to_lds(g0, g1, z4, z4, 0);
#endif
}
#endif

// Stage a (rows x 64) bf16 tile into LDS. TDM path: wave 0 issues one DMA.
// Fallback: cooperative vector copy by all 256 threads.
static __device__ __forceinline__ void stage_tile(__bf16* dst_lds,
                                                  const __bf16* gsrc,
                                                  unsigned tile_rows,
                                                  unsigned tensor_rows,
                                                  unsigned ld_elts) {
#if HAVE_TDM
  if (threadIdx.x < 32)
    tdm_load_tile(lds_offset(dst_lds), gsrc, tile_rows, tensor_rows, ld_elts);
#else
  const int t = threadIdx.x;  // 256 threads, 2 per row
  for (unsigned r = (unsigned)(t >> 1); r < tile_rows; r += 128) {
    const int col = (t & 1) << 5;
    *(v16bf*)(dst_lds + r * 64 + col) =
        *(const v16bf*)(gsrc + (size_t)r * ld_elts + col);
    *(v16bf*)(dst_lds + r * 64 + col + 16) =
        *(const v16bf*)(gsrc + (size_t)r * ld_elts + col + 16);
  }
#endif
}

template <int N>
static __device__ __forceinline__ void wait_tiles() {
#if HAVE_TDM
  __builtin_amdgcn_s_wait_tensorcnt((short)N);  // harmless if counter already 0
#endif
}

// ---------------------------------------------------------------------------
// fp32 -> bf16 conversion
// ---------------------------------------------------------------------------
__global__ void mha_cvt_bf16(const float* __restrict__ in,
                             __bf16* __restrict__ out, int n) {
  int i = blockIdx.x * blockDim.x + threadIdx.x;
  if (i < n) out[i] = (__bf16)in[i];
}

// ---------------------------------------------------------------------------
// QKV GEMM: qkv[m,e] = sum_d x[m,d]*w_qkv[e,d]; M=4096, N=3072, K=1024.
// WG = 8 waves, macro tile 128x128, wave tile 32x64 (2x4 WMMA tiles).
// K staged 64 at a time by TDM into double-buffered LDS (64 KB).
// Epilogue scatters into Q[b,h,s,dh], K[b,h,s,dh], Vt[b,h,dh,s] (bf16).
// ---------------------------------------------------------------------------
__global__ __launch_bounds__(256) void mha_qkv_gemm(
    const __bf16* __restrict__ X, const __bf16* __restrict__ W,
    __bf16* __restrict__ Qb, __bf16* __restrict__ Kb, __bf16* __restrict__ Vt) {
  __shared__ __align__(128) __bf16 ldsA[2][128][64];   // 32 KB
  __shared__ __align__(128) __bf16 ldsB[2][128][64];   // 32 KB

  const int lane = threadIdx.x & 31;
  const int wave = threadIdx.x >> 5;
  const int wm = wave & 3;   // 4 waves along M
  const int wn = wave >> 2;  // 2 waves along N
  const int m0 = blockIdx.x * 128 + wm * 32;
  const int n0 = blockIdx.y * 128 + wn * 64;
  const __bf16* Abase = X + (size_t)(blockIdx.x * 128) * D_;
  const __bf16* Bbase = W + (size_t)(blockIdx.y * 128) * D_;

  constexpr int NS = D_ / 64;  // 16 K-steps
  v8f acc[2][4] = {};

  stage_tile(&ldsA[0][0][0], Abase, 128, MTOT, D_);
  stage_tile(&ldsB[0][0][0], Bbase, 128, E3, D_);

  for (int s = 0; s < NS; ++s) {
    const int cur = s & 1;
    if (s + 1 < NS) {  // prefetch next K-step while this one computes
      stage_tile(&ldsA[cur ^ 1][0][0], Abase + (s + 1) * 64, 128, MTOT, D_);
      stage_tile(&ldsB[cur ^ 1][0][0], Bbase + (s + 1) * 64, 128, E3, D_);
      wait_tiles<2>();  // pair for step s has landed
    } else {
      wait_tiles<0>();
    }
    __syncthreads();

#pragma unroll
    for (int kl = 0; kl < 64; kl += 32) {
      v16bf a0 = load_frag(&ldsA[cur][wm * 32][kl], 64, lane);
      v16bf a1 = load_frag(&ldsA[cur][wm * 32 + 16][kl], 64, lane);
#pragma unroll
      for (int j = 0; j < 4; ++j) {
        v16bf bf = load_frag(&ldsB[cur][wn * 64 + j * 16][kl], 64, lane);
        acc[0][j] = wmma_bf16(a0, bf, acc[0][j]);
        acc[1][j] = wmma_bf16(a1, bf, acc[1][j]);
      }
    }
    __syncthreads();  // done reading buf[cur] before it is refilled
  }

  // C layout: element (row = v + 8*(lane>>4), col = lane&15) per 16x16 tile
#pragma unroll
  for (int i = 0; i < 2; ++i) {
#pragma unroll
    for (int j = 0; j < 4; ++j) {
      const int n = n0 + j * 16 + (lane & 15);
#pragma unroll
      for (int v = 0; v < 8; ++v) {
        const int m = m0 + i * 16 + ((lane >> 4) << 3) + v;
        const int bb = m >> 11;
        const int sq = m & (S_ - 1);
        const float val = acc[i][j][v];
        if (n < D_) {
          const int h = n >> 6, dh = n & 63;
          Qb[(((size_t)(bb * H_ + h)) * S_ + sq) * DH_ + dh] = (__bf16)val;
        } else if (n < 2 * D_) {
          const int e = n - D_;
          const int h = e >> 6, dh = e & 63;
          Kb[(((size_t)(bb * H_ + h)) * S_ + sq) * DH_ + dh] = (__bf16)val;
        } else {
          const int e = n - 2 * D_;
          const int h = e >> 6, dh = e & 63;
          Vt[(((size_t)(bb * H_ + h)) * DH_ + dh) * S_ + sq] = (__bf16)val;
        }
      }
    }
  }
}

// ---------------------------------------------------------------------------
// Flash attention (causal). Grid: (b*H, S/64). WG = 4 waves (128 thr),
// each wave owns 16 q rows. Online softmax, P*V via per-wave LDS tile + Vt.
// ---------------------------------------------------------------------------
__global__ __launch_bounds__(128) void mha_attn(
    const __bf16* __restrict__ Qb, const __bf16* __restrict__ Kb,
    const __bf16* __restrict__ Vt, __bf16* __restrict__ Ob) {
  __shared__ __align__(32) __bf16 plds[4][16][64];  // per-wave P tile, 8 KB

  const int lane = threadIdx.x & 31;
  const int wave = threadIdx.x >> 5;
  const int bh = blockIdx.x;
  const int qb = blockIdx.y;
  const int b = bh >> 4, h = bh & 15;

  const __bf16* Qh = Qb + (size_t)bh * S_ * DH_;
  const __bf16* Kh = Kb + (size_t)bh * S_ * DH_;
  const __bf16* Vh = Vt + (size_t)bh * DH_ * S_;

  const int q0 = qb * 64 + wave * 16;
  const int rowloc = (lane >> 4) << 3;

  const v16bf qf0 = load_frag(Qh + (size_t)q0 * DH_ + 0, DH_, lane);
  const v16bf qf1 = load_frag(Qh + (size_t)q0 * DH_ + 32, DH_, lane);

  v8f oacc[4] = {};
  float mrow[8], lrow[8];
#pragma unroll
  for (int v = 0; v < 8; ++v) { mrow[v] = -1e30f; lrow[v] = 0.0f; }
  const float scale = 0.125f;  // 1/sqrt(64)

  for (int kb = 0; kb <= qb; ++kb) {  // uniform over WG -> barriers safe
    const int key0 = kb * 64;

    if (kb < qb) {  // prefetch next key block (K rows + Vt columns)
      const __bf16* nk = Kh + (size_t)(key0 + 64) * DH_;
      __builtin_prefetch(nk + (size_t)lane * DH_, 0, 1);
      __builtin_prefetch(nk + (size_t)(lane + 32) * DH_, 0, 1);
      __builtin_prefetch(Vh + (size_t)lane * S_ + key0 + 64, 0, 1);
      __builtin_prefetch(Vh + (size_t)(lane + 32) * S_ + key0 + 64, 0, 1);
    }

    // S = Q K^T : 16 q x 64 keys, contraction 64 (2 WMMA steps)
    v8f sc[4];
#pragma unroll
    for (int n = 0; n < 4; ++n) {
      v8f z = {};
      v16bf k0 = load_frag(Kh + (size_t)(key0 + n * 16) * DH_ + 0, DH_, lane);
      v16bf k1 = load_frag(Kh + (size_t)(key0 + n * 16) * DH_ + 32, DH_, lane);
      sc[n] = wmma_bf16(qf0, k0, z);
      sc[n] = wmma_bf16(qf1, k1, sc[n]);
    }

    // causal mask + online softmax; write P (bf16) into this wave's LDS tile
#pragma unroll
    for (int v = 0; v < 8; ++v) {
      const int qg = q0 + rowloc + v;
      float rm = -1e30f;
#pragma unroll
      for (int n = 0; n < 4; ++n) {
        const int kg = key0 + n * 16 + (lane & 15);
        float x = sc[n][v] * scale;
        x = (kg <= qg) ? x : -1e30f;
        sc[n][v] = x;
        rm = fmaxf(rm, x);
      }
#pragma unroll
      for (int off = 1; off < 16; off <<= 1)
        rm = fmaxf(rm, __shfl_xor(rm, off, 32));
      const float mnew = fmaxf(mrow[v], rm);
      const float alpha = __expf(mrow[v] - mnew);
      mrow[v] = mnew;
      float rs = 0.0f;
#pragma unroll
      for (int n = 0; n < 4; ++n) {
        const float p = __expf(sc[n][v] - mnew);
        rs += p;
        plds[wave][rowloc + v][n * 16 + (lane & 15)] = (__bf16)p;
      }
#pragma unroll
      for (int off = 1; off < 16; off <<= 1)
        rs += __shfl_xor(rs, off, 32);
      lrow[v] = lrow[v] * alpha + rs;
#pragma unroll
      for (int n = 0; n < 4; ++n) oacc[n][v] = oacc[n][v] * alpha;
    }
    __syncthreads();

    // O += P V : contraction = 64 keys (2 steps), N = 64 dh (4 tiles)
#pragma unroll
    for (int half = 0; half < 2; ++half) {
      v16bf pa = load_frag(&plds[wave][0][0] + half * 32, 64, lane);
#pragma unroll
      for (int n = 0; n < 4; ++n) {
        v16bf vb =
            load_frag(Vh + (size_t)(n * 16) * S_ + key0 + half * 32, S_, lane);
        oacc[n] = wmma_bf16(pa, vb, oacc[n]);
      }
    }
    __syncthreads();
  }

#pragma unroll
  for (int v = 0; v < 8; ++v) {
    const float inv = 1.0f / lrow[v];
    const int srow = q0 + rowloc + v;
#pragma unroll
    for (int n = 0; n < 4; ++n) {
      const int col = h * DH_ + n * 16 + (lane & 15);
      Ob[((size_t)b * S_ + srow) * D_ + col] = (__bf16)(oacc[n][v] * inv);
    }
  }
}

// ---------------------------------------------------------------------------
// Output projection: out[m,e] = sum_d O[m,d]*w_proj[e,d]; fp32 out.
// Same TDM-staged tiling; M=4096, N=1024, K=1024.
// ---------------------------------------------------------------------------
__global__ __launch_bounds__(256) void mha_proj_gemm(
    const __bf16* __restrict__ A, const __bf16* __restrict__ W,
    float* __restrict__ out) {
  __shared__ __align__(128) __bf16 ldsA[2][128][64];
  __shared__ __align__(128) __bf16 ldsB[2][128][64];

  const int lane = threadIdx.x & 31;
  const int wave = threadIdx.x >> 5;
  const int wm = wave & 3;
  const int wn = wave >> 2;
  const int m0 = blockIdx.x * 128 + wm * 32;
  const int n0 = blockIdx.y * 128 + wn * 64;
  const __bf16* Abase = A + (size_t)(blockIdx.x * 128) * D_;
  const __bf16* Bbase = W + (size_t)(blockIdx.y * 128) * D_;

  constexpr int NS = D_ / 64;
  v8f acc[2][4] = {};

  stage_tile(&ldsA[0][0][0], Abase, 128, MTOT, D_);
  stage_tile(&ldsB[0][0][0], Bbase, 128, D_, D_);

  for (int s = 0; s < NS; ++s) {
    const int cur = s & 1;
    if (s + 1 < NS) {
      stage_tile(&ldsA[cur ^ 1][0][0], Abase + (s + 1) * 64, 128, MTOT, D_);
      stage_tile(&ldsB[cur ^ 1][0][0], Bbase + (s + 1) * 64, 128, D_, D_);
      wait_tiles<2>();
    } else {
      wait_tiles<0>();
    }
    __syncthreads();

#pragma unroll
    for (int kl = 0; kl < 64; kl += 32) {
      v16bf a0 = load_frag(&ldsA[cur][wm * 32][kl], 64, lane);
      v16bf a1 = load_frag(&ldsA[cur][wm * 32 + 16][kl], 64, lane);
#pragma unroll
      for (int j = 0; j < 4; ++j) {
        v16bf bf = load_frag(&ldsB[cur][wn * 64 + j * 16][kl], 64, lane);
        acc[0][j] = wmma_bf16(a0, bf, acc[0][j]);
        acc[1][j] = wmma_bf16(a1, bf, acc[1][j]);
      }
    }
    __syncthreads();
  }

#pragma unroll
  for (int i = 0; i < 2; ++i) {
#pragma unroll
    for (int j = 0; j < 4; ++j) {
      const int n = n0 + j * 16 + (lane & 15);
#pragma unroll
      for (int v = 0; v < 8; ++v) {
        const int m = m0 + i * 16 + ((lane >> 4) << 3) + v;
        out[(size_t)m * D_ + n] = acc[i][j][v];
      }
    }
  }
}

// ---------------------------------------------------------------------------
// Launch
// ---------------------------------------------------------------------------
extern "C" void kernel_launch(void* const* d_in, const int* in_sizes, int n_in,
                              void* d_out, int out_size, void* d_ws,
                              size_t ws_size, hipStream_t stream) {
  (void)in_sizes; (void)n_in; (void)out_size; (void)ws_size;
  const float* x = (const float*)d_in[0];
  const float* w_qkv = (const float*)d_in[1];
  const float* w_proj = (const float*)d_in[2];
  float* out = (float*)d_out;

  char* ws = (char*)d_ws;
  size_t off = 0;
  auto salloc = [&](size_t bytes) {
    void* p = ws + off;
    off += (bytes + 255) & ~(size_t)255;
    return p;
  };
  __bf16* xb     = (__bf16*)salloc((size_t)MTOT * D_ * 2);
  __bf16* wqkvb  = (__bf16*)salloc((size_t)E3 * D_ * 2);
  __bf16* wprojb = (__bf16*)salloc((size_t)D_ * D_ * 2);
  __bf16* Qb     = (__bf16*)salloc((size_t)MTOT * D_ * 2);
  __bf16* Kb     = (__bf16*)salloc((size_t)MTOT * D_ * 2);
  __bf16* Vt     = (__bf16*)salloc((size_t)MTOT * D_ * 2);
  __bf16* Ob     = (__bf16*)salloc((size_t)MTOT * D_ * 2);

  const int nx = MTOT * D_;
  const int nw = E3 * D_;
  const int np = D_ * D_;
  mha_cvt_bf16<<<(nx + 255) / 256, 256, 0, stream>>>(x, xb, nx);
  mha_cvt_bf16<<<(nw + 255) / 256, 256, 0, stream>>>(w_qkv, wqkvb, nw);
  mha_cvt_bf16<<<(np + 255) / 256, 256, 0, stream>>>(w_proj, wprojb, np);

  mha_qkv_gemm<<<dim3(MTOT / 128, E3 / 128), 256, 0, stream>>>(xb, wqkvb, Qb,
                                                               Kb, Vt);
  mha_attn<<<dim3(B_ * H_, S_ / 64), 128, 0, stream>>>(Qb, Kb, Vt, Ob);
  mha_proj_gemm<<<dim3(MTOT / 128, D_ / 128), 256, 0, stream>>>(Ob, wprojb,
                                                                out);
}
